// CAPAtt_46480136077364
// MI455X (gfx1250) — compile-verified, matrix operand order checked by reading, not run on hardware
//
#include <hip/hip_runtime.h>
#include <hip/hip_bf16.h>
#include <math.h>

typedef __attribute__((ext_vector_type(16))) __bf16 v16bf;
typedef __attribute__((ext_vector_type(8)))  float  v8f;

#define BM 128
#define BN 64
#define BK 32
#define LDSA 36   // padded fp32 LDS row stride: 144B rows -> 16-row bank spread, 16B aligned
#define NBUF 3    // async pipeline depth 2 -> 3 LDS buffers

__device__ __forceinline__ float sigmoidf_(float x) {
    return 1.0f / (1.0f + __expf(-x));
}

// low 32 bits of a flat pointer into the LDS aperture == LDS byte offset (ISA 10.2)
__device__ __forceinline__ unsigned lds_addr(const void* p) {
    return (unsigned)(size_t)p;
}

__device__ __forceinline__ void async_load_b128(unsigned lds, const float* gptr) {
    asm volatile("global_load_async_to_lds_b128 %0, %1, off"
                 :: "v"(lds), "v"(gptr) : "memory");
}

// wait until at most N async-LDS loads outstanding (one staged tile = 6 loads)
__device__ __forceinline__ void wait_async_le(int newer_tiles) {
    if (newer_tiles >= 1) asm volatile("s_wait_asynccnt 0x6" ::: "memory");
    else                  asm volatile("s_wait_asynccnt 0x0" ::: "memory");
}

// ---------------------------------------------------------------------------
// C[M,N] (+)= A[M,K] @ W[N,K]^T + bias0 + bias1   (fp32 in/out, bf16 WMMA)
// Block: 256 threads (8 wave32). Tile: 128x64, K-step 32.
// Triple-buffered LDS fed by global_load_async_to_lds_b128 (ASYNCcnt),
// prefetch distance 2; partial s_wait_asynccnt so tile i+1 stays in flight.
// Wave w: wm = w&3 -> 32 rows, wn = w>>2 -> 32 cols; 2x2 wmma 16x16x32 frags.
// ---------------------------------------------------------------------------
__global__ __launch_bounds__(256)
void gemm_bf16_wmma(const float* __restrict__ A, const float* __restrict__ W,
                    const float* __restrict__ bias0, const float* __restrict__ bias1,
                    float* __restrict__ C, int M, int N, int K, int accumulate)
{
    __shared__ float As[NBUF][BM][LDSA];
    __shared__ float Bs[NBUF][BN][LDSA];

    const int tid    = threadIdx.x;
    const int lane   = tid & 31;
    const int wave   = tid >> 5;
    const int wm     = wave & 3;        // 0..3 : rows  [wm*32, wm*32+32)
    const int wn     = wave >> 2;       // 0..1 : cols  [wn*32, wn*32+32)
    const int lane_r = lane & 15;
    const int khalf  = (lane >> 4) * 8; // K sub-half per ISA bf16 A/B layout

    const int m0 = blockIdx.y * BM;
    const int n0 = blockIdx.x * BN;

    // Per-thread staging chunk coordinates (loop-invariant).
    // A tile: 128 rows x 32 floats = 1024 float4 chunks, 4 per thread.
    // W tile:  64 rows x 32 floats =  512 float4 chunks, 2 per thread.
    int rA[4], cA[4], rW[2], cW[2];
    #pragma unroll
    for (int it = 0; it < 4; ++it) {
        const int c = tid + it * 256;
        rA[it] = c >> 3;
        cA[it] = (c & 7) * 4;
    }
    #pragma unroll
    for (int it = 0; it < 2; ++it) {
        const int c = tid + it * 256;
        rW[it] = c >> 3;
        cW[it] = (c & 7) * 4;
    }

    auto stage = [&](int p, int k0) {   // 6 async b128 loads per wave-lane
        #pragma unroll
        for (int it = 0; it < 4; ++it)
            async_load_b128(lds_addr(&As[p][rA[it]][cA[it]]),
                            &A[(size_t)(m0 + rA[it]) * K + k0 + cA[it]]);
        #pragma unroll
        for (int it = 0; it < 2; ++it)
            async_load_b128(lds_addr(&Bs[p][rW[it]][cW[it]]),
                            &W[(size_t)(n0 + rW[it]) * K + k0 + cW[it]]);
    };

    v8f acc[2][2] = {};

    // pipeline prologue: tiles 0 and 1 in flight
    stage(0, 0);
    if (BK < K) stage(1, BK);

    int iter = 0;
    for (int k0 = 0; k0 < K; k0 += BK, ++iter) {
        const int p = iter % NBUF;

        // tile i landed (my 6 loads); tile i+1's 6 may still be in flight
        wait_async_le((k0 + BK < K) ? 1 : 0);
        // everyone's tile-i chunks landed; everyone done reading tile i-1
        __syncthreads();
        // stage tile i+2 into the buffer tile i-1 occupied (now safe)
        if (k0 + 2 * BK < K) stage((iter + 2) % NBUF, k0 + 2 * BK);

        // ---- hoist all 16 LDS b128 reads, then convert (batched dscnt) ----
        // lane l<16 : row lane_r, K {0..7,16..23};  l>=16 : same row, K {8..15,24..31}
        float4 qa[2][4], qb[2][4];
        #pragma unroll
        for (int i = 0; i < 2; ++i) {
            const float4* q = reinterpret_cast<const float4*>(
                &As[p][wm * 32 + i * 16 + lane_r][khalf]);
            qa[i][0] = q[0]; qa[i][1] = q[1]; qa[i][2] = q[4]; qa[i][3] = q[5];
        }
        #pragma unroll
        for (int i = 0; i < 2; ++i) {
            const float4* q = reinterpret_cast<const float4*>(
                &Bs[p][wn * 32 + i * 16 + lane_r][khalf]);
            qb[i][0] = q[0]; qb[i][1] = q[1]; qb[i][2] = q[4]; qb[i][3] = q[5];
        }

        v16bf a[2], b[2];
        #pragma unroll
        for (int i = 0; i < 2; ++i) {
            #pragma unroll
            for (int c = 0; c < 4; ++c) {
                a[i][4 * c + 0] = (__bf16)qa[i][c].x;
                a[i][4 * c + 1] = (__bf16)qa[i][c].y;
                a[i][4 * c + 2] = (__bf16)qa[i][c].z;
                a[i][4 * c + 3] = (__bf16)qa[i][c].w;
                b[i][4 * c + 0] = (__bf16)qb[i][c].x;
                b[i][4 * c + 1] = (__bf16)qb[i][c].y;
                b[i][4 * c + 2] = (__bf16)qb[i][c].z;
                b[i][4 * c + 3] = (__bf16)qb[i][c].w;
            }
        }

        #pragma unroll
        for (int i = 0; i < 2; ++i)
            #pragma unroll
            for (int j = 0; j < 2; ++j)
                acc[i][j] = __builtin_amdgcn_wmma_f32_16x16x32_bf16(
                    false, a[i], false, b[j], (short)0, acc[i][j], false, false);
    }

    // ---- epilogue: C/D layout — lane holds col (lane&15); VGPR r -> row
    //      r + 8*(lane>=16). Fuse bias add and optional accumulate. ----
    const int mrow = khalf;
    #pragma unroll
    for (int j = 0; j < 2; ++j) {
        const int n = n0 + wn * 32 + j * 16 + lane_r;
        float bsum = 0.0f;
        if (bias0) bsum += bias0[n];
        if (bias1) bsum += bias1[n];
        #pragma unroll
        for (int i = 0; i < 2; ++i) {
            #pragma unroll
            for (int r = 0; r < 8; ++r) {
                const int m = m0 + wm * 32 + i * 16 + mrow + r;
                const size_t idx = (size_t)m * N + n;
                const float val = acc[i][j][r] + bsum;
                if (accumulate) C[idx] += val;
                else            C[idx]  = val;
            }
        }
    }
}

// ---------------------------------------------------------------------------
// x1[b,e] = relu(embed_W[ids[b], e])     (B=256 blocks, E=1024)
// ---------------------------------------------------------------------------
__global__ __launch_bounds__(256)
void embed_relu_kernel(const float* __restrict__ embedW, const int* __restrict__ ids,
                       float* __restrict__ x1)
{
    const int b  = blockIdx.x;
    const size_t base = (size_t)ids[b] * 1024;
    for (int h = threadIdx.x; h < 1024; h += 256)
        x1[(size_t)b * 1024 + h] = fmaxf(embedW[base + h], 0.0f);
}

// ---------------------------------------------------------------------------
// LSTM cell 1: gates[B,4H] -> h1, c1
// ---------------------------------------------------------------------------
__global__ __launch_bounds__(256)
void lstm1_kernel(const float* __restrict__ gates, const float* __restrict__ c1_0,
                  float* __restrict__ h1, float* __restrict__ c1)
{
    const int idx = blockIdx.x * 256 + threadIdx.x;  // b*1024 + h
    const int b = idx >> 10;
    const int h = idx & 1023;
    const float* g = &gates[(size_t)b * 4096];
    const float i  = sigmoidf_(g[h]);
    const float f  = sigmoidf_(g[1024 + h]);
    const float gg = tanhf(g[2048 + h]);
    const float o  = sigmoidf_(g[3072 + h]);
    const float c  = f * c1_0[idx] + i * gg;
    c1[idx] = c;
    h1[idx] = o * tanhf(c);
}

// ---------------------------------------------------------------------------
// Attention: alpha[b,n] = whW . tanh(WvV[b,n,:] + wghc[b,:]) + whb
//            att = softmax_n(alpha); av[b,:] = sum_n att[n] * v[b,n,:]
// One block per b; 8 waves each reduce rows n = wave, wave+8, ...
// ---------------------------------------------------------------------------
__global__ __launch_bounds__(256)
void attention_kernel(const float* __restrict__ v, const float* __restrict__ WvV,
                      const float* __restrict__ wghc, const float* __restrict__ whW,
                      const float* __restrict__ whb, float* __restrict__ av)
{
    const int b    = blockIdx.x;
    const int tid  = threadIdx.x;
    const int lane = tid & 31;
    const int wave = tid >> 5;

    __shared__ float s_g[1024];
    __shared__ float s_att[64];

    for (int h = tid; h < 1024; h += 256) s_g[h] = wghc[(size_t)b * 1024 + h];
    __syncthreads();

    for (int n = wave; n < 36; n += 8) {
        const float* row = &WvV[((size_t)b * 36 + n) * 1024];
        float partial = 0.0f;
        for (int h = lane; h < 1024; h += 32)
            partial += whW[h] * tanhf(row[h] + s_g[h]);
        #pragma unroll
        for (int off = 16; off > 0; off >>= 1)
            partial += __shfl_xor(partial, off, 32);
        if (lane == 0) s_att[n] = partial + whb[0];
    }
    __syncthreads();

    // softmax over 36 entries (computed redundantly; trivially cheap)
    float mx = -1e30f;
    for (int n = 0; n < 36; ++n) mx = fmaxf(mx, s_att[n]);
    float sum = 0.0f;
    for (int n = 0; n < 36; ++n) sum += __expf(s_att[n] - mx);
    const float inv = 1.0f / sum;
    __syncthreads();
    if (tid < 36) s_att[tid] = __expf(s_att[tid] - mx) * inv;
    __syncthreads();

    for (int h = tid; h < 1024; h += 256) {
        float acc = 0.0f;
        for (int n = 0; n < 36; ++n)
            acc += s_att[n] * v[((size_t)b * 36 + n) * 1024 + h];
        av[(size_t)b * 1024 + h] = acc;
    }
}

// ---------------------------------------------------------------------------
// LSTM cell 2: ig[B,4H], th[B,2H] -> h2, c2
// o_t = sig(ig[2H:3H]+th[0:H]); f_t = sig(ig[3H:4H]+th[H:2H])
// ---------------------------------------------------------------------------
__global__ __launch_bounds__(256)
void lstm2_kernel(const float* __restrict__ ig, const float* __restrict__ th,
                  const float* __restrict__ c2_0, float* __restrict__ h2,
                  float* __restrict__ c2)
{
    const int idx = blockIdx.x * 256 + threadIdx.x;
    const int b = idx >> 10;
    const int h = idx & 1023;
    const float* g = &ig[(size_t)b * 4096];
    const float* t = &th[(size_t)b * 2048];
    const float i_t = sigmoidf_(g[h]);
    const float g_t = tanhf(g[1024 + h]);
    const float o_t = sigmoidf_(g[2048 + h] + t[h]);
    const float f_t = sigmoidf_(g[3072 + h] + t[1024 + h]);
    const float c   = g_t * i_t + c2_0[idx] * f_t;
    c2[idx] = c;
    h2[idx] = tanhf(c) * o_t;
}

// ---------------------------------------------------------------------------
extern "C" void kernel_launch(void* const* d_in, const int* in_sizes, int n_in,
                              void* d_out, int out_size, void* d_ws, size_t ws_size,
                              hipStream_t stream)
{
    const float* v      = (const float*)d_in[0];
    const float* h1_0   = (const float*)d_in[1];
    const float* c1_0   = (const float*)d_in[2];
    const float* h2_0   = (const float*)d_in[3];
    const float* c2_0   = (const float*)d_in[4];
    const int*   ids    = (const int*)  d_in[5];
    const float* embedW = (const float*)d_in[6];
    const float* Wih    = (const float*)d_in[7];
    const float* Whh    = (const float*)d_in[8];
    const float* bih    = (const float*)d_in[9];
    const float* bhh    = (const float*)d_in[10];
    const float* wvW    = (const float*)d_in[11];
    const float* wvb    = (const float*)d_in[12];
    const float* wgW    = (const float*)d_in[13];
    const float* wgb    = (const float*)d_in[14];
    const float* wcW    = (const float*)d_in[15];
    const float* wcb    = (const float*)d_in[16];
    const float* whW    = (const float*)d_in[17];
    const float* whb    = (const float*)d_in[18];
    const float* i2hW   = (const float*)d_in[19];
    const float* i2hb   = (const float*)d_in[20];
    const float* h2hW   = (const float*)d_in[21];
    const float* h2hb   = (const float*)d_in[22];
    const float* t2hW   = (const float*)d_in[23];
    const float* t2hb   = (const float*)d_in[24];
    const float* mlpW   = (const float*)d_in[25];
    const float* mlpb   = (const float*)d_in[26];

    // outputs: (output[256,32000], h1, c1, h2, c2) flat in order
    float* out = (float*)d_out;
    float* h1  = out + (size_t)256 * 32000;
    float* c1  = h1 + 256 * 1024;
    float* h2  = c1 + 256 * 1024;
    float* c2  = h2 + 256 * 1024;

    // workspace layout (floats)
    float* ws    = (float*)d_ws;
    float* x1    = ws;                          // 256*1024
    float* gates = x1 + 256 * 1024;             // 256*4096
    float* WvV   = gates + 256 * 4096;          // 9216*1024
    float* wghc  = WvV + (size_t)9216 * 1024;   // 256*1024
    float* av    = wghc + 256 * 1024;           // 256*1024
    float* ig    = av + 256 * 1024;             // 256*4096
    float* th    = ig + 256 * 4096;             // 256*2048

    const dim3 blk(256);

    // 1) embedding gather + relu
    embed_relu_kernel<<<256, blk, 0, stream>>>(embedW, ids, x1);

    // 2) LSTM-1 gates = x1@Wih^T + bih + bhh  (then += h1_0@Whh^T)
    gemm_bf16_wmma<<<dim3(4096 / BN, 256 / BM), blk, 0, stream>>>(
        x1, Wih, bih, bhh, gates, 256, 4096, 1024, 0);
    gemm_bf16_wmma<<<dim3(4096 / BN, 256 / BM), blk, 0, stream>>>(
        h1_0, Whh, nullptr, nullptr, gates, 256, 4096, 1024, 1);
    lstm1_kernel<<<1024, blk, 0, stream>>>(gates, c1_0, h1, c1);

    // 3) attention pre-projections
    gemm_bf16_wmma<<<dim3(1024 / BN, 9216 / BM), blk, 0, stream>>>(
        v, wvW, wvb, nullptr, WvV, 9216, 1024, 1024, 0);
    gemm_bf16_wmma<<<dim3(1024 / BN, 256 / BM), blk, 0, stream>>>(
        h1, wgW, wgb, nullptr, wghc, 256, 1024, 1024, 0);
    gemm_bf16_wmma<<<dim3(1024 / BN, 256 / BM), blk, 0, stream>>>(
        c2_0, wcW, wcb, nullptr, wghc, 256, 1024, 1024, 1);

    // 4) attention scores + softmax + weighted sum
    attention_kernel<<<256, blk, 0, stream>>>(v, WvV, wghc, whW, whb, av);

    // 5) LSTM-2 gates
    gemm_bf16_wmma<<<dim3(4096 / BN, 256 / BM), blk, 0, stream>>>(
        av, i2hW, i2hb, h2hb, ig, 256, 4096, 1024, 0);
    gemm_bf16_wmma<<<dim3(4096 / BN, 256 / BM), blk, 0, stream>>>(
        h2_0, h2hW, nullptr, nullptr, ig, 256, 4096, 1024, 1);
    gemm_bf16_wmma<<<dim3(2048 / BN, 256 / BM), blk, 0, stream>>>(
        h1, t2hW, t2hb, nullptr, th, 256, 2048, 1024, 0);
    lstm2_kernel<<<1024, blk, 0, stream>>>(ig, th, c2_0, h2, c2);

    // 6) vocab projection (memory-bound: 131 MB of mlp_W streamed once)
    gemm_bf16_wmma<<<dim3(32000 / BN, 256 / BM), blk, 0, stream>>>(
        h2, mlpW, mlpb, nullptr, out, 256, 32000, 1024, 0);
}
